// CSDGI_EnDecoder_45526653337682
// MI455X (gfx1250) — matrix operand order repaired and checked by reading, not compiled, not run on hardware
//
#include <hip/hip_runtime.h>
#include <hip/hip_bf16.h>

typedef __attribute__((ext_vector_type(16))) _Float16 v16h;
typedef __attribute__((ext_vector_type(8)))  _Float16 v8h;
typedef __attribute__((ext_vector_type(8)))  float    v8f;

#define MDIM   1024
#define BDIM   512
#define PALL   96
#define NROWS  (BDIM * PALL)   // 49152 GEMM rows
#define LRANK  64
#define KS     64              // K per LDS stage (2 WMMA k-steps)
#define LDP    (KS + 8)        // padded LDS row stride (144B -> 36 banks)

// ---------------------------------------------------------------------------
// 1) normalize gw rows: wn[r][:] = gw[r][:] / max(||gw[r]||, eps)
// ---------------------------------------------------------------------------
__global__ __launch_bounds__(256)
void gw_norm_kernel(const float* __restrict__ gw, float* __restrict__ wn) {
    __shared__ float red[256];
    const int r = blockIdx.x;            // 0..63
    const int t = threadIdx.x;
    float s = 0.f;
    for (int i = t; i < MDIM; i += 256) { float v = gw[r * MDIM + i]; s += v * v; }
    red[t] = s; __syncthreads();
    for (int st = 128; st > 0; st >>= 1) {
        if (t < st) red[t] += red[t + st];
        __syncthreads();
    }
    const float inv = 1.0f / fmaxf(sqrtf(red[0]), 1e-10f);
    for (int i = t; i < MDIM; i += 256) wn[r * MDIM + i] = gw[r * MDIM + i] * inv;
}

// ---------------------------------------------------------------------------
// 2) graw[m][n] = | sum_k wn[k][m] * wn[k][n] |
// ---------------------------------------------------------------------------
__global__ __launch_bounds__(256)
void gram_kernel(const float* __restrict__ wn, float* __restrict__ graw) {
    __shared__ float wm[LRANK];
    const int n = blockIdx.x * 256 + threadIdx.x;   // grid.x = 4
    const int m = blockIdx.y;                       // grid.y = 1024
    if (threadIdx.x < LRANK) wm[threadIdx.x] = wn[threadIdx.x * MDIM + m];
    __syncthreads();
    float s = 0.f;
#pragma unroll 8
    for (int k = 0; k < LRANK; ++k) s += wm[k] * wn[k * MDIM + n];
    graw[(size_t)m * MDIM + n] = fabsf(s);
}

// ---------------------------------------------------------------------------
// 3a) rowsum[m] = sum_n graw[m][n]
// ---------------------------------------------------------------------------
__global__ __launch_bounds__(256)
void rowsum_kernel(const float* __restrict__ graw, float* __restrict__ rowsum) {
    __shared__ float red[256];
    const int m = blockIdx.x;
    const int t = threadIdx.x;
    float s = 0.f;
    for (int n = t; n < MDIM; n += 256) s += graw[(size_t)m * MDIM + n];
    red[t] = s; __syncthreads();
    for (int st = 128; st > 0; st >>= 1) {
        if (t < st) red[t] += red[t + st];
        __syncthreads();
    }
    if (t == 0) rowsum[m] = red[0];
}

// ---------------------------------------------------------------------------
// 3b) gT[n][m] = f16( (m==n) ? 0 : graw[m][n] / rowsum[m] )   (transposed!)
// ---------------------------------------------------------------------------
__global__ __launch_bounds__(256)
void gnorm_kernel(const float* __restrict__ graw, const float* __restrict__ rowsum,
                  _Float16* __restrict__ gT) {
    const int m = blockIdx.x * 256 + threadIdx.x;   // grid.x = 4
    const int n = blockIdx.y;                       // grid.y = 1024
    float v = graw[(size_t)m * MDIM + n] / rowsum[m];
    if (m == n) v = 0.f;
    gT[(size_t)n * MDIM + m] = (_Float16)v;         // coalesced over m
}

// ---------------------------------------------------------------------------
// 4) encoder: residual Linear stack -> x_sp f16 in [b][p][m] layout
// ---------------------------------------------------------------------------
__global__ __launch_bounds__(256)
void encoder_kernel(const float* __restrict__ in,
                    const float* __restrict__ w0, const float* __restrict__ b0,
                    const float* __restrict__ w1, const float* __restrict__ b1,
                    const float* __restrict__ w2, const float* __restrict__ b2,
                    _Float16* __restrict__ xsp) {
    __shared__ float sw1[32 * 32], sw2[32 * 32];
    __shared__ float sw0[32], sb0[32], sb1[32], sb2[32];
    const int t = threadIdx.x;
    for (int i = t; i < 1024; i += 256) { sw1[i] = w1[i]; sw2[i] = w2[i]; }
    if (t < 32) { sw0[t] = w0[t]; sb0[t] = b0[t]; sb1[t] = b1[t]; sb2[t] = b2[t]; }
    __syncthreads();

    const int m = blockIdx.x * 256 + t;   // grid.x = 4
    const int b = blockIdx.y;             // grid.y = 512
    const float x0 = in[(size_t)b * MDIM + m];

    float h0[32], h1[32], h2[32];
#pragma unroll
    for (int o = 0; o < 32; ++o) h0[o] = fmaxf(sw0[o] * x0 + sb0[o] + x0, 0.f);
#pragma unroll 4
    for (int o = 0; o < 32; ++o) {
        float s = sb1[o];
#pragma unroll
        for (int i = 0; i < 32; ++i) s += sw1[o * 32 + i] * h0[i];
        h1[o] = fmaxf(s + h0[o], 0.f);
    }
#pragma unroll 4
    for (int o = 0; o < 32; ++o) {
        float s = sb2[o];
#pragma unroll
        for (int i = 0; i < 32; ++i) s += sw2[o * 32 + i] * h1[i];
        h2[o] = fmaxf(s + h1[o], 0.f);
    }
    const size_t base = ((size_t)b * PALL) * MDIM + m;
#pragma unroll
    for (int p = 0; p < 32; ++p) xsp[base + (size_t)p * MDIM]        = (_Float16)h0[p];
#pragma unroll
    for (int p = 0; p < 32; ++p) xsp[base + (size_t)(p + 32) * MDIM] = (_Float16)h1[p];
#pragma unroll
    for (int p = 0; p < 32; ++p) xsp[base + (size_t)(p + 64) * MDIM] = (_Float16)h2[p];
}

// ---------------------------------------------------------------------------
// 5) init outputs with fb (outputs get atomic contributions from the GEMM)
// ---------------------------------------------------------------------------
__global__ __launch_bounds__(256)
void init_out_kernel(const float* __restrict__ fb0, const float* __restrict__ fb1,
                     float* __restrict__ y) {
    const int idx = blockIdx.x * 256 + threadIdx.x;   // 2*512*1024 total
    const int c = idx >> 19;                          // 512*1024 = 2^19
    const int n = idx & (MDIM - 1);
    y[idx] = (c == 0 ? fb0 : fb1)[n];
}

// ---------------------------------------------------------------------------
// fragment helpers (CDNA5 wave32 WMMA f16 layouts)
// ---------------------------------------------------------------------------
__device__ inline v16h frag_a(const _Float16* row, int hs) {
    // A 16x32: lane<16 -> K {0..7, 16..23}; lane>=16 -> K {8..15, 24..31}
    v8h lo = *(const v8h*)(row + hs * 8);
    v8h hi = *(const v8h*)(row + 16 + hs * 8);
    v16h r;
#pragma unroll
    for (int i = 0; i < 8; ++i) { r[i] = lo[i]; r[i + 8] = hi[i]; }
    return r;
}
__device__ inline v16h frag_b(const _Float16* col, int hs) {
    // B 32x16 (from gT row): lane<16 -> K {0..7, 16..23}; lane>=16 -> K {8..15, 24..31}
    v8h lo = *(const v8h*)(col + hs * 8);
    v8h hi = *(const v8h*)(col + 16 + hs * 8);
    v16h r;
#pragma unroll
    for (int i = 0; i < 8; ++i) { r[i] = lo[i]; r[i + 8] = hi[i]; }
    return r;
}

// ---------------------------------------------------------------------------
// 6) main WMMA GEMM + fused tanh/FR epilogue (software-pipelined)
//    C[row, n] = sum_m xsp[row, m] * g[m, n] ; row = b*96 + p
//    y[b, n] += tanh((C + gb[n]) * 0.1) * fw[n*96 + p]
// ---------------------------------------------------------------------------
__global__ __launch_bounds__(256)
void gemm_kernel(const _Float16* __restrict__ A,   // [NROWS][1024]
                 const _Float16* __restrict__ Bt,  // gT: [1024 n][1024 m]
                 const float* __restrict__ gb,
                 const float* __restrict__ fw,     // [1024][96]
                 float* __restrict__ y) {          // [512][1024]
    __shared__ _Float16 As[128][LDP];   // 128 x 72 halves = 18 KB
    __shared__ _Float16 Bs[128][LDP];

    const int tid  = threadIdx.x;
    const int lane = tid & 31;
    const int wave = tid >> 5;          // 8 waves
    const int wr   = wave & 3;          // 4 row groups * 32 rows
    const int wc   = wave >> 2;         // 2 col groups * 64 cols
    const int hs   = lane >> 4;         // lane half select
    const int lrow = lane & 15;

    const int row0 = blockIdx.y * 128;  // grid.y = 384
    const int col0 = blockIdx.x * 128;  // grid.x = 8

    // chunk mapping for stage load/store: 1024 chunks of 8 halves
    const int cr0 = (tid >> 3);         // row for chunk j=0 .. (+32 per j)
    const int cseg = (tid & 7) * 8;     // half offset within row

    v8f acc[2][4];
#pragma unroll
    for (int i = 0; i < 2; ++i)
#pragma unroll
        for (int j = 0; j < 4; ++j) acc[i][j] = (v8f){0.f,0.f,0.f,0.f,0.f,0.f,0.f,0.f};

    v8h pa[4], pb[4];
    // prefetch stage 0
#pragma unroll
    for (int j = 0; j < 4; ++j) {
        const int r = cr0 + j * 32;
        pa[j] = *(const v8h*)&A [(size_t)(row0 + r) * MDIM + cseg];
        pb[j] = *(const v8h*)&Bt[(size_t)(col0 + r) * MDIM + cseg];
    }

    for (int kt = 0; kt < MDIM / KS; ++kt) {
        __syncthreads();
#pragma unroll
        for (int j = 0; j < 4; ++j) {
            const int r = cr0 + j * 32;
            *(v8h*)&As[r][cseg] = pa[j];
            *(v8h*)&Bs[r][cseg] = pb[j];
        }
        __syncthreads();

        if (kt + 1 < MDIM / KS) {       // prefetch next stage during math
            const int k0 = (kt + 1) * KS;
#pragma unroll
            for (int j = 0; j < 4; ++j) {
                const int r = cr0 + j * 32;
                pa[j] = *(const v8h*)&A [(size_t)(row0 + r) * MDIM + k0 + cseg];
                pb[j] = *(const v8h*)&Bt[(size_t)(col0 + r) * MDIM + k0 + cseg];
            }
        }

#pragma unroll
        for (int ks = 0; ks < 2; ++ks) {                 // two WMMA k-steps
            v16h af[2], bf[4];
#pragma unroll
            for (int sr = 0; sr < 2; ++sr)
                af[sr] = frag_a(&As[wr * 32 + sr * 16 + lrow][ks * 32], hs);
#pragma unroll
            for (int sc = 0; sc < 4; ++sc)
                bf[sc] = frag_b(&Bs[wc * 64 + sc * 16 + lrow][ks * 32], hs);
#pragma unroll
            for (int sr = 0; sr < 2; ++sr)
#pragma unroll
                for (int sc = 0; sc < 4; ++sc)
                    acc[sr][sc] = __builtin_amdgcn_wmma_f32_16x16x32_f16(
                        false, af[sr], false, bf[sc], (short)0, acc[sr][sc], false, false);
        }
    }

    // fused epilogue: every 16-row D tile lies within one batch b (96 = 6*16),
    // so reduce the 8 per-lane elements in registers -> one atomic per tile.
#pragma unroll
    for (int sr = 0; sr < 2; ++sr) {
        const int rbase = row0 + wr * 32 + sr * 16;   // wave-uniform
        const int b     = rbase / PALL;               // wave-uniform
        const int pbase = rbase - b * PALL;
#pragma unroll
        for (int sc = 0; sc < 4; ++sc) {
            const int n = col0 + wc * 64 + sc * 16 + lrow;
            const float gbn = gb[n];
            const float* fwn = &fw[(size_t)n * PALL + pbase + 8 * hs];
            float s = 0.f;
#pragma unroll
            for (int j = 0; j < 8; ++j)
                s += tanhf((acc[sr][sc][j] + gbn) * 0.1f) * fwn[j];
            atomicAdd(&y[(size_t)b * MDIM + n], s);
        }
    }
}

// ---------------------------------------------------------------------------
extern "C" void kernel_launch(void* const* d_in, const int* in_sizes, int n_in,
                              void* d_out, int out_size, void* d_ws, size_t ws_size,
                              hipStream_t stream) {
    const float* in_x = (const float*)d_in[0];
    const float* w0 = (const float*)d_in[1];
    const float* b0 = (const float*)d_in[2];
    const float* w1 = (const float*)d_in[3];
    const float* b1 = (const float*)d_in[4];
    const float* w2 = (const float*)d_in[5];
    const float* b2 = (const float*)d_in[6];

    // workspace layout
    uint8_t* ws = (uint8_t*)d_ws;
    _Float16* xsp = (_Float16*)ws;                 ws += (size_t)NROWS * MDIM * 2;  // 96 MB
    float* wn     = (float*)ws;                    ws += (size_t)LRANK * MDIM * 4;
    float* graw   = (float*)ws;                    ws += (size_t)MDIM * MDIM * 4;   // 4 MB
    float* rowsum = (float*)ws;                    ws += 4096;
    _Float16* gT  = (_Float16*)ws;                 ws += (size_t)MDIM * MDIM * 2;   // 2 MB

    float* y = (float*)d_out;

    encoder_kernel<<<dim3(4, BDIM), 256, 0, stream>>>(in_x, w0, b0, w1, b1, w2, b2, xsp);
    init_out_kernel<<<(2 * BDIM * MDIM) / 256, 256, 0, stream>>>(
        (const float*)d_in[10], (const float*)d_in[14], y);

    for (int c = 0; c < 2; ++c) {
        const float* gw = (const float*)d_in[7 + 4 * c];
        const float* gbp = (const float*)d_in[8 + 4 * c];
        const float* fw = (const float*)d_in[9 + 4 * c];
        float* yc = y + (size_t)c * BDIM * MDIM;

        gw_norm_kernel<<<LRANK, 256, 0, stream>>>(gw, wn);
        gram_kernel<<<dim3(4, MDIM), 256, 0, stream>>>(wn, graw);
        rowsum_kernel<<<MDIM, 256, 0, stream>>>(graw, rowsum);
        gnorm_kernel<<<dim3(4, MDIM), 256, 0, stream>>>(graw, rowsum, gT);
        gemm_kernel<<<dim3(MDIM / 128, NROWS / 128), 256, 0, stream>>>(xsp, gT, gbp, fw, yc);
    }
}
